// CrossAttention_4131758539031
// MI455X (gfx1250) — matrix operand order repaired and checked by reading, not compiled, not run on hardware
//
#include <hip/hip_runtime.h>
#include <hip/hip_bf16.h>
#include <stdint.h>

// Problem constants (match reference setup_inputs)
#define EMB   1024
#define HEADS 16
#define HD    64
#define BATCH 2
#define TX    1024
#define TC    8192

typedef __attribute__((ext_vector_type(16))) __bf16 v16bf;
typedef __attribute__((ext_vector_type(8)))  float  v8f;
typedef __attribute__((ext_vector_type(4)))  unsigned int v4u;
typedef __attribute__((ext_vector_type(8)))  unsigned int v8u;

union Frag {
    unsigned int u[8];
    v16bf v;
};

__device__ __forceinline__ unsigned short f2bf(float f) {
    union { float f; unsigned int u; } c; c.f = f;
    unsigned int u = c.u;
    unsigned int r = (u + 0x7fffu + ((u >> 16) & 1u)) >> 16; // RNE
    return (unsigned short)r;
}

__device__ __forceinline__ v8f wmma_bf16(const Frag& a, const Frag& b, v8f c) {
    return __builtin_amdgcn_wmma_f32_16x16x32_bf16(
        /*neg_a=*/false, a.v, /*neg_b=*/false, b.v,
        /*c_mod=*/(short)0, c, /*reuse_a=*/false, /*reuse_b=*/false);
}

__device__ __forceinline__ v8f zero8() {
    v8f z = {0.f,0.f,0.f,0.f,0.f,0.f,0.f,0.f};
    return z;
}

// ---------------------------------------------------------------------------
// Tensor Data Mover: 2D tile load global -> LDS with row padding.
// D# layout per CDNA5 ISA ch.8:
//   group0: [1:0]=count=1, [63:32]=lds_addr, [120:64]=global_addr, [127:126]=2
//   group1: [17:16]=data_size(3 => 8B), [20]=pad_en, [24:22]=pad_interval,
//           [31:25]=pad_amount, [79:48]=tensor_dim0, [111:80]=tensor_dim1,
//           [127:112]=tile_dim0, [143:128]=tile_dim1, [159:144]=tile_dim2=0,
//           [207:160]=tensor_dim0_stride
//   groups 2/3: zero (2D tile).
// Units for dims/strides: 8-byte elements (data_size code 3).
// ---------------------------------------------------------------------------
__device__ __forceinline__ void tdm_load_2d(
    unsigned lds_byte_off, const void* gaddr,
    unsigned tensor_d0, unsigned tensor_d1,
    unsigned d0_stride,
    unsigned tile_d0, unsigned tile_d1,
    unsigned pad_interval_code, unsigned pad_amount_code)
{
    const unsigned long long ga = (unsigned long long)gaddr;
    v4u g0;
    g0[0] = 0x1u;                                   // count=1, user descriptor
    g0[1] = lds_byte_off;                           // lds_addr
    g0[2] = (unsigned)(ga & 0xffffffffu);           // global_addr[31:0]
    g0[3] = (unsigned)((ga >> 32) & 0x01ffffffu)    // global_addr[56:32]
            | (0x2u << 30);                         // type = 2 ("image")
    v8u g1;
    g1[0] = (3u << 16)                              // data_size = 8B
            | (1u << 20)                            // pad_enable
            | (pad_interval_code << 22)
            | (pad_amount_code << 25);
    g1[1] = (tensor_d0 & 0xffffu) << 16;
    g1[2] = (tensor_d0 >> 16) | ((tensor_d1 & 0xffffu) << 16);
    g1[3] = (tensor_d1 >> 16) | (tile_d0 << 16);
    g1[4] = tile_d1;                                // tile_dim2 = 0 (2D)
    g1[5] = d0_stride;                              // stride[31:0]
    g1[6] = 0;                                      // stride[47:32], dim1_stride lo
    g1[7] = 0;
    const v4u gz = {0u, 0u, 0u, 0u};
    asm volatile("tensor_load_to_lds %0, %1, %2, %3"
                 :: "s"(g0), "s"(g1), "s"(gz), "s"(gz)
                 : "memory");
}

__device__ __forceinline__ unsigned lds_off(const void* p) {
    return (unsigned)(unsigned long long)p;   // flat->LDS: addr[31:0]
}

// ---------------------------------------------------------------------------
// fp32 -> bf16 elementwise (weights)
// ---------------------------------------------------------------------------
__global__ __launch_bounds__(256) void f32_to_bf16_kernel(
    const float* __restrict__ src, unsigned short* __restrict__ dst, int n)
{
    int i = (blockIdx.x * 256 + threadIdx.x) * 4;
    if (i + 3 < n) {
        float4 v = *(const float4*)(src + i);
        uint2 o;
        o.x = (unsigned int)f2bf(v.x) | ((unsigned int)f2bf(v.y) << 16);
        o.y = (unsigned int)f2bf(v.z) | ((unsigned int)f2bf(v.w) << 16);
        *(uint2*)(dst + i) = o;
    }
}

// ---------------------------------------------------------------------------
// LayerNorm (one row of EMB per workgroup), fp32 in -> bf16 out
// ---------------------------------------------------------------------------
__global__ __launch_bounds__(256) void layernorm_to_bf16(
    const float* __restrict__ x, const float* __restrict__ g,
    const float* __restrict__ b, unsigned short* __restrict__ out)
{
    const int row = blockIdx.x;
    const int tid = threadIdx.x;
    const float4 v = *(const float4*)(x + (size_t)row * EMB + tid * 4);
    float s = v.x + v.y + v.z + v.w;
    float q = v.x*v.x + v.y*v.y + v.z*v.z + v.w*v.w;
    #pragma unroll
    for (int m = 16; m >= 1; m >>= 1) {
        s += __shfl_xor(s, m, 32);
        q += __shfl_xor(q, m, 32);
    }
    __shared__ float ss[8], sq[8];
    if ((tid & 31) == 0) { ss[tid >> 5] = s; sq[tid >> 5] = q; }
    __syncthreads();
    s = 0.f; q = 0.f;
    #pragma unroll
    for (int i = 0; i < 8; ++i) { s += ss[i]; q += sq[i]; }
    const float mean = s * (1.0f / EMB);
    const float var  = q * (1.0f / EMB) - mean * mean;
    const float rstd = rsqrtf(var + 1e-5f);
    const int c = tid * 4;
    const float4 gv = *(const float4*)(g + c);
    const float4 bv = *(const float4*)(b + c);
    float o0 = (v.x - mean) * rstd * gv.x + bv.x;
    float o1 = (v.y - mean) * rstd * gv.y + bv.y;
    float o2 = (v.z - mean) * rstd * gv.z + bv.z;
    float o3 = (v.w - mean) * rstd * gv.w + bv.w;
    uint2 o;
    o.x = (unsigned int)f2bf(o0) | ((unsigned int)f2bf(o1) << 16);
    o.y = (unsigned int)f2bf(o2) | ((unsigned int)f2bf(o3) << 16);
    *(uint2*)(out + (size_t)row * EMB + c) = o;
}

// ---------------------------------------------------------------------------
// NT GEMM: C[M,N] = A[M,K] (bf16, row-major) x W[N,K]^T (bf16, row-major)
// Block tile 128x128, BK=32, 8 waves; each wave 32x64 = 2x4 WMMA tiles.
// A/W tiles staged by TDM (double buffered): 128 rows x 64B, +16B LDS pad
// per row -> 80B row stride (As[.][.][40]).
// FP32_OUT=0 -> bf16 C; FP32_OUT=1 -> f32 C (+ optional bias).
// ---------------------------------------------------------------------------
template <int FP32_OUT>
__global__ __launch_bounds__(256) void gemm_nt_bf16(
    const unsigned short* __restrict__ A,
    const unsigned short* __restrict__ W,
    unsigned short* __restrict__ Cb,
    float* __restrict__ Cf,
    const float* __restrict__ bias,
    int M, int N, int K)
{
    __shared__ __align__(16) unsigned short As[2][128][40];
    __shared__ __align__(16) unsigned short Bs[2][128][40];

    const int tid  = threadIdx.x;
    const int lane = tid & 31;
    const int wv   = tid >> 5;
    const int wm   = wv >> 1;   // 0..3  (rows)
    const int wn   = wv & 1;    // 0..1  (cols)
    const int r    = lane & 15;
    const int hi   = lane >> 4;
    const int bn   = blockIdx.x;
    const int bm   = blockIdx.y;

    const unsigned t_d0 = (unsigned)(K / 4);  // row length in 8B units
    // tile: 8 x 128 (8B units x rows); pad: every 16 DWORDs add 4 DWORDs
    auto issue = [&](int buf, int kt) {
        tdm_load_2d(lds_off(&As[buf][0][0]),
                    A + (size_t)(bm * 128) * K + kt,
                    t_d0, (unsigned)M, t_d0, 8u, 128u, 3u, 3u);
        tdm_load_2d(lds_off(&Bs[buf][0][0]),
                    W + (size_t)(bn * 128) * K + kt,
                    t_d0, (unsigned)N, t_d0, 8u, 128u, 3u, 3u);
    };

    v8f acc[2][4];
    #pragma unroll
    for (int ms = 0; ms < 2; ++ms)
        #pragma unroll
        for (int ns = 0; ns < 4; ++ns)
            acc[ms][ns] = zero8();

    if (wv == 0) issue(0, 0);

    int buf = 0;
    for (int kt = 0; kt < K; kt += 32, buf ^= 1) {
        const bool has_next = (kt + 32) < K;
        if (wv == 0) {
            if (has_next) {
                issue(buf ^ 1, kt + 32);
                __builtin_amdgcn_s_wait_tensorcnt(2);  // current tile landed
            } else {
                __builtin_amdgcn_s_wait_tensorcnt(0);
            }
        }
        __syncthreads();

        Frag af[2];
        #pragma unroll
        for (int ms = 0; ms < 2; ++ms) {
            const int row = wm * 32 + ms * 16 + r;
            #pragma unroll
            for (int v = 0; v < 8; ++v) {
                const int k0 = 2 * v + 8 * hi + ((v >= 4) ? 8 : 0);
                af[ms].u[v] = *(const unsigned int*)&As[buf][row][k0];
            }
        }
        #pragma unroll
        for (int ns = 0; ns < 4; ++ns) {
            Frag bf;
            const int col = wn * 64 + ns * 16 + r;
            #pragma unroll
            for (int v = 0; v < 8; ++v)
                bf.u[v] = *(const unsigned int*)&Bs[buf][col][2 * v + 8 * hi + ((v >= 4) ? 8 : 0)];
            #pragma unroll
            for (int ms = 0; ms < 2; ++ms)
                acc[ms][ns] = wmma_bf16(af[ms], bf, acc[ms][ns]);
        }
        __syncthreads();
    }

    // Epilogue: C layout -> element j lives at row (j + 8*hi), col (lane&15)
    #pragma unroll
    for (int ms = 0; ms < 2; ++ms) {
        #pragma unroll
        for (int ns = 0; ns < 4; ++ns) {
            #pragma unroll
            for (int j = 0; j < 8; ++j) {
                const int row = bm * 128 + wm * 32 + ms * 16 + j + 8 * hi;
                const int col = bn * 128 + wn * 64 + ns * 16 + r;
                const float val = acc[ms][ns][j];
                if (FP32_OUT)
                    Cf[(size_t)row * N + col] = val + (bias ? bias[col] : 0.0f);
                else
                    Cb[(size_t)row * N + col] = f2bf(val);
            }
        }
    }
}

// ---------------------------------------------------------------------------
// Flash attention: one workgroup per (batch, head, 64-query tile).
// 4 waves, each owns 16 query rows. Online softmax over Tc in 64-key tiles.
// K/V tiles staged by TDM (double buffered): 64 rows x 128B, +16B pad per
// row -> 144B row stride (Kt[.][.][72]).
// ---------------------------------------------------------------------------
__global__ __launch_bounds__(128) void flash_attn(
    const unsigned short* __restrict__ Q,   // [B*TX, EMB] bf16
    const unsigned short* __restrict__ K,   // [B*TC, EMB] bf16
    const unsigned short* __restrict__ V,   // [B*TC, EMB] bf16
    const unsigned char*  __restrict__ mask,// [B, TC] bool
    unsigned short* __restrict__ O)         // [B*TX, EMB] bf16
{
    __shared__ __align__(16) unsigned short Kt[2][64][72];
    __shared__ __align__(16) unsigned short Vt[2][64][72];
    __shared__ __align__(16) unsigned short Pw[4][16][72];

    const int b  = blockIdx.z;
    const int h  = blockIdx.y;
    const int qt = blockIdx.x;
    const int tid  = threadIdx.x;
    const int lane = tid & 31;
    const int wv   = tid >> 5;
    const int r    = lane & 15;
    const int hi   = lane >> 4;

    // row length in 8B units for the [B*T, EMB] bf16 matrices
    const unsigned t_d0 = EMB / 4;
    // tile: 16 x 64 (8B units x rows); pad: every 32 DWORDs add 4 DWORDs
    auto issue = [&](int buf, int k0g) {
        const size_t base = (size_t)(b * TC + k0g) * EMB + h * HD;
        tdm_load_2d(lds_off(&Kt[buf][0][0]), K + base,
                    t_d0, (unsigned)(BATCH * TC), t_d0, 16u, 64u, 4u, 3u);
        tdm_load_2d(lds_off(&Vt[buf][0][0]), V + base,
                    t_d0, (unsigned)(BATCH * TC), t_d0, 16u, 64u, 4u, 3u);
    };

    // Q A-fragments: 16 rows x 64 dims = two 16x32 fragments, kept in regs
    Frag qf[2];
    {
        const unsigned short* qp =
            Q + (size_t)(b * TX + qt * 64 + wv * 16 + r) * EMB + h * HD;
        #pragma unroll
        for (int f = 0; f < 2; ++f) {
            #pragma unroll
            for (int v = 0; v < 8; ++v) {
                const int k0 = f * 32 + 2 * v + 8 * hi + ((v >= 4) ? 8 : 0);
                qf[f].u[v] = *(const unsigned int*)(qp + k0);
            }
        }
    }

    float m_run[8], l_run[8];
    v8f o[4];
    #pragma unroll
    for (int j = 0; j < 8; ++j) { m_run[j] = -INFINITY; l_run[j] = 0.f; }
    #pragma unroll
    for (int n = 0; n < 4; ++n) o[n] = zero8();

    if (wv == 0) issue(0, 0);

    int buf = 0;
    for (int k0g = 0; k0g < TC; k0g += 64, buf ^= 1) {
        const bool has_next = (k0g + 64) < TC;
        if (wv == 0) {
            if (has_next) {
                issue(buf ^ 1, k0g + 64);
                __builtin_amdgcn_s_wait_tensorcnt(2);  // current K/V landed
            } else {
                __builtin_amdgcn_s_wait_tensorcnt(0);
            }
        }
        __syncthreads();

        // S = (Q x K^T) / sqrt(HD), 4 tiles of 16 keys
        v8f s[4];
        #pragma unroll
        for (int kt = 0; kt < 4; ++kt) {
            v8f a = zero8();
            #pragma unroll
            for (int f = 0; f < 2; ++f) {
                Frag kb;  // B-fragment: K-dim pairs contiguous in Kt row
                #pragma unroll
                for (int v = 0; v < 8; ++v)
                    kb.u[v] = *(const unsigned int*)
                        &Kt[buf][kt * 16 + r][f * 32 + 2 * v + 8 * hi + ((v >= 4) ? 8 : 0)];
                a = wmma_bf16(qf[f], kb, a);
            }
            const float msk =
                mask[(size_t)b * TC + k0g + kt * 16 + r] ? 0.0f : -1e30f;
            #pragma unroll
            for (int j = 0; j < 8; ++j)
                s[kt][j] = a[j] * 0.125f + msk;
        }

        // Online softmax statistics (row = j + 8*hi, cols across 16 lanes)
        #pragma unroll
        for (int j = 0; j < 8; ++j) {
            float t = fmaxf(fmaxf(s[0][j], s[1][j]), fmaxf(s[2][j], s[3][j]));
            #pragma unroll
            for (int m = 8; m >= 1; m >>= 1)
                t = fmaxf(t, __shfl_xor(t, m, 32));
            const float mn    = fmaxf(m_run[j], t);
            const float alpha = __expf(m_run[j] - mn);
            float rs = 0.f;
            #pragma unroll
            for (int kt = 0; kt < 4; ++kt) {
                const float p = __expf(s[kt][j] - mn);
                s[kt][j] = p;
                rs += p;
            }
            #pragma unroll
            for (int m = 8; m >= 1; m >>= 1)
                rs += __shfl_xor(rs, m, 32);
            l_run[j] = l_run[j] * alpha + rs;
            m_run[j] = mn;
            #pragma unroll
            for (int n = 0; n < 4; ++n)
                o[n][j] *= alpha;
        }

        // P (C layout) -> LDS -> re-read as A-fragments (per-wave region)
        #pragma unroll
        for (int kt = 0; kt < 4; ++kt)
            #pragma unroll
            for (int j = 0; j < 8; ++j)
                Pw[wv][j + 8 * hi][kt * 16 + r] = f2bf(s[kt][j]);
        asm volatile("s_wait_dscnt 0" ::: "memory");

        // O += P x V
        #pragma unroll
        for (int f = 0; f < 2; ++f) {
            Frag pf;
            #pragma unroll
            for (int v = 0; v < 8; ++v) {
                const int k0 = f * 32 + 2 * v + 8 * hi + ((v >= 4) ? 8 : 0);
                pf.u[v] = *(const unsigned int*)&Pw[wv][r][k0];
            }
            #pragma unroll
            for (int n = 0; n < 4; ++n) {
                Frag vf;  // B-fragment: K pairs are consecutive keys (strided rows)
                #pragma unroll
                for (int v = 0; v < 8; ++v) {
                    const int kk = f * 32 + 2 * v + 8 * hi + ((v >= 4) ? 8 : 0);
                    const unsigned int lo = Vt[buf][kk][n * 16 + r];
                    const unsigned int h2 = Vt[buf][kk + 1][n * 16 + r];
                    vf.u[v] = lo | (h2 << 16);
                }
                o[n] = wmma_bf16(pf, vf, o[n]);
            }
        }
        __syncthreads();
    }

    // Normalize and store (bf16) into [B*TX, EMB] at head slice
    float inv[8];
    #pragma unroll
    for (int j = 0; j < 8; ++j)
        inv[j] = (l_run[j] > 0.f) ? (1.0f / l_run[j]) : 0.0f;
    #pragma unroll
    for (int n = 0; n < 4; ++n) {
        #pragma unroll
        for (int j = 0; j < 8; ++j) {
            const int row = qt * 64 + wv * 16 + j + 8 * hi;
            O[(size_t)(b * TX + row) * EMB + h * HD + n * 16 + r] =
                f2bf(o[n][j] * inv[j]);
        }
    }
}

// ---------------------------------------------------------------------------
// Host-side launch
// ---------------------------------------------------------------------------
extern "C" void kernel_launch(void* const* d_in, const int* in_sizes, int n_in,
                              void* d_out, int out_size, void* d_ws, size_t ws_size,
                              hipStream_t stream) {
    (void)in_sizes; (void)n_in; (void)out_size; (void)ws_size;
    const float* x    = (const float*)d_in[0];
    const float* ctx  = (const float*)d_in[1];
    const unsigned char* mask = (const unsigned char*)d_in[2];
    const float* Wq = (const float*)d_in[3];
    const float* Wk = (const float*)d_in[4];
    const float* Wv = (const float*)d_in[5];
    const float* Wo = (const float*)d_in[6];
    const float* bo = (const float*)d_in[7];
    const float* g1 = (const float*)d_in[8];
    const float* b1 = (const float*)d_in[9];
    const float* g2 = (const float*)d_in[10];
    const float* b2 = (const float*)d_in[11];
    float* out = (float*)d_out;

    // Workspace layout (bf16 intermediates), ~116 MiB total
    char* ws = (char*)d_ws;
    size_t off = 0;
    auto alloc = [&](size_t bytes) {
        char* p = ws + off;
        off += (bytes + 255) & ~(size_t)255;
        return (unsigned short*)p;
    };
    const size_t WN = (size_t)EMB * EMB;
    unsigned short* Wqb = alloc(WN * 2);
    unsigned short* Wkb = alloc(WN * 2);
    unsigned short* Wvb = alloc(WN * 2);
    unsigned short* Wob = alloc(WN * 2);
    unsigned short* XN  = alloc((size_t)BATCH * TX * EMB * 2);
    unsigned short* CN  = alloc((size_t)BATCH * TC * EMB * 2);
    unsigned short* Qb  = alloc((size_t)BATCH * TX * EMB * 2);
    unsigned short* Kb  = alloc((size_t)BATCH * TC * EMB * 2);
    unsigned short* Vb  = alloc((size_t)BATCH * TC * EMB * 2);
    unsigned short* AO  = alloc((size_t)BATCH * TX * EMB * 2);

    // 1) weights fp32 -> bf16
    const int cvtBlocks = (int)(WN / 1024);
    f32_to_bf16_kernel<<<cvtBlocks, 256, 0, stream>>>(Wq, Wqb, (int)WN);
    f32_to_bf16_kernel<<<cvtBlocks, 256, 0, stream>>>(Wk, Wkb, (int)WN);
    f32_to_bf16_kernel<<<cvtBlocks, 256, 0, stream>>>(Wv, Wvb, (int)WN);
    f32_to_bf16_kernel<<<cvtBlocks, 256, 0, stream>>>(Wo, Wob, (int)WN);

    // 2) layernorms
    layernorm_to_bf16<<<BATCH * TX, 256, 0, stream>>>(x, g1, b1, XN);
    layernorm_to_bf16<<<BATCH * TC, 256, 0, stream>>>(ctx, g2, b2, CN);

    // 3) Q/K/V projections (bf16 out)
    dim3 gq(EMB / 128, (BATCH * TX) / 128);
    dim3 gk(EMB / 128, (BATCH * TC) / 128);
    gemm_nt_bf16<0><<<gq, 256, 0, stream>>>(XN, Wqb, Qb, nullptr, nullptr,
                                            BATCH * TX, EMB, EMB);
    gemm_nt_bf16<0><<<gk, 256, 0, stream>>>(CN, Wkb, Kb, nullptr, nullptr,
                                            BATCH * TC, EMB, EMB);
    gemm_nt_bf16<0><<<gk, 256, 0, stream>>>(CN, Wvb, Vb, nullptr, nullptr,
                                            BATCH * TC, EMB, EMB);

    // 4) attention
    dim3 ga(TX / 64, HEADS, BATCH);
    flash_attn<<<ga, 128, 0, stream>>>(Qb, Kb, Vb, mask, AO);

    // 5) output projection (f32 out + bias)
    gemm_nt_bf16<1><<<gq, 256, 0, stream>>>(AO, Wob, nullptr, out, bo,
                                            BATCH * TX, EMB, EMB);
}
